// LanguageQuantizer_72911364817042
// MI455X (gfx1250) — matrix operand order repaired and checked by reading, not compiled, not run on hardware
//
#include <hip/hip_runtime.h>
#include <math.h>

typedef __attribute__((ext_vector_type(16))) __bf16 v16bf;
typedef __attribute__((ext_vector_type(8)))  __bf16 bf16x8;
typedef __attribute__((ext_vector_type(8)))  float  v8f;

#define N_TOK 8192
#define DIM   1024
#define LAT   512
#define VCB   8192

// ---------------- fragment loaders (ISA 7.12.2 layouts, wave32) ----------------

// A-matrix 16x32 bf16: lanes 0-15 row r hold K {0..7}U{16..23}; lanes 16-31 K {8..15}U{24..31}
__device__ __forceinline__ v16bf load_a_bf16(const __bf16* __restrict__ A, int ld,
                                             int row0, int k0, int lane) {
  const int r = lane & 15, h = lane >> 4;
  const __bf16* p = A + (size_t)(row0 + r) * ld + k0 + h * 8;
  bf16x8 lo = *(const bf16x8*)p;
  bf16x8 hi = *(const bf16x8*)(p + 16);
  v16bf f;
#pragma unroll
  for (int i = 0; i < 8; ++i) { f[i] = lo[i]; f[i + 8] = hi[i]; }
  return f;
}

// Same A layout but source is f32 (convert to bf16 on the fly)
__device__ __forceinline__ v16bf load_a_f32(const float* __restrict__ A, int ld,
                                            int row0, int k0, int lane) {
  const int r = lane & 15, h = lane >> 4;
  const float* p = A + (size_t)(row0 + r) * ld + k0 + h * 8;
  v16bf f;
#pragma unroll
  for (int i = 0; i < 8; ++i) { f[i] = (__bf16)p[i]; f[i + 8] = (__bf16)p[16 + i]; }
  return f;
}

// B-matrix 32x16 bf16 from a K-contiguous BT[ncols][K]: lanes 0-15 col n hold K {0..7}U{16..23},
// lanes 16-31 hold K {8..15}U{24..31}  -> two 16-byte loads per lane.
__device__ __forceinline__ v16bf load_b_bf16(const __bf16* __restrict__ BT, int ld,
                                             int col0, int k0, int lane) {
  const int n = lane & 15, h = lane >> 4;
  const __bf16* p = BT + (size_t)(col0 + n) * ld + k0 + h * 8;
  bf16x8 lo = *(const bf16x8*)p;
  bf16x8 hi = *(const bf16x8*)(p + 16);
  v16bf f;
#pragma unroll
  for (int i = 0; i < 8; ++i) { f[i] = lo[i]; f[i + 8] = hi[i]; }
  return f;
}

__device__ __forceinline__ unsigned int orderable_f32(float v) {
  unsigned int b = __float_as_uint(v);
  return (b & 0x80000000u) ? ~b : (b | 0x80000000u);
}

// ---------------- prep: transpose W [1024,512] f32 -> WT [512,1024] bf16 ----------------
__global__ __launch_bounds__(256) void prep_w_kernel(const float* __restrict__ W,
                                                     __bf16* __restrict__ WT) {
  int gid = blockIdx.x * 256 + threadIdx.x;      // 0 .. 512*1024-1
  int l = gid >> 10;
  int d = gid & 1023;
  WT[gid] = (__bf16)W[(size_t)d * LAT + l];
}

__global__ __launch_bounds__(256) void init_kernel(unsigned long long* __restrict__ packed,
                                                   int* __restrict__ hist,
                                                   float* __restrict__ accum) {
  int i = blockIdx.x * 256 + threadIdx.x;
  if (i < VCB) { packed[i] = 0ull; hist[i] = 0; }
  if (i < 2) accum[i] = 0.0f;
}

// ---------------- GEMM: P = A[M,1024](f32) * WT^T + bias ;  M = 8192 ----------------
// mode 0: store P to outP[M,512]; mode 1: sse += (P - cmpP)^2 -> atomicAdd(accum)
__global__ __launch_bounds__(128) void gemm_latent_kernel(
    const float* __restrict__ A, const __bf16* __restrict__ WT,
    const float* __restrict__ bias, float* __restrict__ outP,
    const float* __restrict__ cmpP, float* __restrict__ accum, int mode) {
  const int lane = threadIdx.x & 31;
  const int wave = blockIdx.x * 4 + (threadIdx.x >> 5);   // 512 waves
  const int row0 = wave * 16;
  const int r = lane & 15, h = lane >> 4;

  v16bf a[32];
#pragma unroll
  for (int kf = 0; kf < 32; ++kf) a[kf] = load_a_f32(A, DIM, row0, kf * 32, lane);

  float sse = 0.0f;
  for (int n0 = 0; n0 < LAT; n0 += 16) {
    v8f acc = {0.f, 0.f, 0.f, 0.f, 0.f, 0.f, 0.f, 0.f};
#pragma unroll
    for (int kf = 0; kf < 32; ++kf) {
      v16bf b = load_b_bf16(WT, DIM, n0, kf * 32, lane);
      acc = __builtin_amdgcn_wmma_f32_16x16x32_bf16(false, a[kf], false, b,
                                                    (short)0, acc, false, false);
    }
    const int col = n0 + r;
    const float bv = bias[col];
    if (mode == 0) {
#pragma unroll
      for (int rr = 0; rr < 8; ++rr) {
        int m = row0 + rr + 8 * h;
        outP[(size_t)m * LAT + col] = acc[rr] + bv;
      }
    } else {
#pragma unroll
      for (int rr = 0; rr < 8; ++rr) {
        int m = row0 + rr + 8 * h;
        float d = acc[rr] + bv - cmpP[(size_t)m * LAT + col];
        sse += d * d;
      }
    }
  }
  if (mode == 1) {
#pragma unroll
    for (int msk = 16; msk >= 1; msk >>= 1) sse += __shfl_xor(sse, msk, 32);
    if (lane == 0) atomicAdd(accum, sse);
  }
}

// ---------------- row l2-normalize: f32 [rows,512] -> bf16 [rows,512] ----------------
__global__ __launch_bounds__(256) void normalize_kernel(const float* __restrict__ P,
                                                        __bf16* __restrict__ out, int rows) {
  int wave = blockIdx.x * 8 + (threadIdx.x >> 5);
  if (wave >= rows) return;
  int lane = threadIdx.x & 31;
  const float* p = P + (size_t)wave * LAT;
  float v[16], s = 0.0f;
#pragma unroll
  for (int i = 0; i < 16; ++i) { v[i] = p[lane + i * 32]; s += v[i] * v[i]; }
#pragma unroll
  for (int m = 16; m >= 1; m >>= 1) s += __shfl_xor(s, m, 32);
  float sc = rsqrtf(s + 1e-12f);
  __bf16* o = out + (size_t)wave * LAT;
#pragma unroll
  for (int i = 0; i < 16; ++i) o[lane + i * 32] = (__bf16)(v[i] * sc);
}

// ---------------- fused similarity GEMM + argmax (unit vectors: argmin dist == argmax dot) ----
// Waves in a block share the same code slice (lock-step B stream -> WGP$/L2 reuse),
// and each iteration handles 2 n-tiles -> 4 independent WMMA accumulator chains.
__global__ __launch_bounds__(128) void dist_argmax_kernel(
    const __bf16* __restrict__ Abf, const __bf16* __restrict__ Cbf,
    unsigned long long* __restrict__ packed) {
  const int lane = threadIdx.x & 31;
  const int wave = blockIdx.x * 4 + (threadIdx.x >> 5);  // 1024 waves
  const int slice = wave >> 8;                           // 0..3 : 2048 codes per slice
  const int tblk  = wave & 255;                          // 32-token block
  const int row0 = tblk * 32;
  const int r = lane & 15, h = lane >> 4;

  v16bf a[2][16];
#pragma unroll
  for (int s = 0; s < 2; ++s)
#pragma unroll
    for (int kf = 0; kf < 16; ++kf)
      a[s][kf] = load_a_bf16(Abf, LAT, row0 + s * 16, kf * 32, lane);

  float bestv[2][8];
  int bestn[2][8];
#pragma unroll
  for (int s = 0; s < 2; ++s)
#pragma unroll
    for (int rr = 0; rr < 8; ++rr) { bestv[s][rr] = -1e30f; bestn[s][rr] = 0; }

  const int nbeg = slice * 2048, nend = nbeg + 2048;
  for (int n0 = nbeg; n0 < nend; n0 += 32) {
    if (n0 + 32 < nend) {
      __builtin_prefetch(Cbf + (size_t)(n0 + 32 + r) * LAT, 0, 1);
      __builtin_prefetch(Cbf + (size_t)(n0 + 48 + r) * LAT, 0, 1);
    }
    v8f acc00 = {0.f, 0.f, 0.f, 0.f, 0.f, 0.f, 0.f, 0.f};  // rows 0-15,  codes n0..n0+15
    v8f acc01 = {0.f, 0.f, 0.f, 0.f, 0.f, 0.f, 0.f, 0.f};  // rows 0-15,  codes n0+16..n0+31
    v8f acc10 = {0.f, 0.f, 0.f, 0.f, 0.f, 0.f, 0.f, 0.f};  // rows 16-31, codes n0..n0+15
    v8f acc11 = {0.f, 0.f, 0.f, 0.f, 0.f, 0.f, 0.f, 0.f};  // rows 16-31, codes n0+16..n0+31
#pragma unroll
    for (int kf = 0; kf < 16; ++kf) {
      v16bf b0 = load_b_bf16(Cbf, LAT, n0,      kf * 32, lane);
      v16bf b1 = load_b_bf16(Cbf, LAT, n0 + 16, kf * 32, lane);
      acc00 = __builtin_amdgcn_wmma_f32_16x16x32_bf16(false, a[0][kf], false, b0,
                                                      (short)0, acc00, false, false);
      acc01 = __builtin_amdgcn_wmma_f32_16x16x32_bf16(false, a[0][kf], false, b1,
                                                      (short)0, acc01, false, false);
      acc10 = __builtin_amdgcn_wmma_f32_16x16x32_bf16(false, a[1][kf], false, b0,
                                                      (short)0, acc10, false, false);
      acc11 = __builtin_amdgcn_wmma_f32_16x16x32_bf16(false, a[1][kf], false, b1,
                                                      (short)0, acc11, false, false);
    }
    const int na = n0 + r;         // this lane's code column, first tile
    const int nb = n0 + 16 + r;    // second tile
#pragma unroll
    for (int rr = 0; rr < 8; ++rr) {
      if (acc00[rr] > bestv[0][rr]) { bestv[0][rr] = acc00[rr]; bestn[0][rr] = na; }
      if (acc01[rr] > bestv[0][rr]) { bestv[0][rr] = acc01[rr]; bestn[0][rr] = nb; }
      if (acc10[rr] > bestv[1][rr]) { bestv[1][rr] = acc10[rr]; bestn[1][rr] = na; }
      if (acc11[rr] > bestv[1][rr]) { bestv[1][rr] = acc11[rr]; bestn[1][rr] = nb; }
    }
  }

  // butterfly reduce across the 16 column-lanes of each half, then one u64 atomic per token
#pragma unroll
  for (int s = 0; s < 2; ++s)
#pragma unroll
    for (int rr = 0; rr < 8; ++rr) {
      float v = bestv[s][rr];
      int n = bestn[s][rr];
#pragma unroll
      for (int m = 1; m <= 8; m <<= 1) {
        float ov = __shfl_xor(v, m, 32);
        int on = __shfl_xor(n, m, 32);
        if (ov > v || (ov == v && on < n)) { v = ov; n = on; }
      }
      if ((lane & 15) == 0) {
        unsigned long long pk =
            ((unsigned long long)orderable_f32(v) << 32) | (unsigned int)(~n);
        int m = row0 + s * 16 + rr + 8 * h;
        atomicMax(&packed[m], pk);
      }
    }
}

// ---------------- gather codebook row, straight-through output, sse1, histogram ----------------
__global__ __launch_bounds__(256) void gather_kernel(
    const unsigned long long* __restrict__ packed, const float* __restrict__ codebook,
    const float* __restrict__ x, float* __restrict__ qout,
    int* __restrict__ hist, float* __restrict__ accum) {
  const int m = blockIdx.x;
  int idx = (int)(~(unsigned int)packed[m]) & (VCB - 1);
  const float* crow = codebook + (size_t)idx * DIM;
  const float* xrow = x + (size_t)m * DIM;
  float* orow = qout + (size_t)m * DIM;
  float s = 0.0f;
  for (int j = threadIdx.x; j < DIM; j += 256) {
    float q = crow[j];
    float xv = xrow[j];
    float d = q - xv;
    orow[j] = xv + d;                // x + (q - x): matches reference rounding
    s += d * d;
  }
  const int lane = threadIdx.x & 31, wid = threadIdx.x >> 5;
#pragma unroll
  for (int msk = 16; msk >= 1; msk >>= 1) s += __shfl_xor(s, msk, 32);
  __shared__ float red[8];
  if (lane == 0) red[wid] = s;
  __syncthreads();
  if (threadIdx.x == 0) {
    float t = 0.0f;
    for (int i = 0; i < 8; ++i) t += red[i];
    atomicAdd(accum + 0, t);
    atomicAdd(hist + idx, 1);
  }
}

// ---------------- final scalars: loss, perplexity, codebook usage ----------------
__global__ __launch_bounds__(256) void finalize_kernel(const int* __restrict__ hist,
                                                       const float* __restrict__ accum,
                                                       float* __restrict__ out) {
  __shared__ float slp[256];
  __shared__ int sus[256];
  float lp = 0.0f;
  int us = 0;
  for (int v = threadIdx.x; v < VCB; v += 256) {
    int c = hist[v];
    if (c > 0) {
      us++;
      float pb = (float)c / (float)N_TOK;
      lp -= pb * logf(pb + 1e-10f);
    }
  }
  slp[threadIdx.x] = lp;
  sus[threadIdx.x] = us;
  __syncthreads();
  if (threadIdx.x == 0) {
    float L = 0.0f;
    int U = 0;
    for (int i = 0; i < 256; ++i) { L += slp[i]; U += sus[i]; }
    float loss = 1.25f * accum[0] / ((float)N_TOK * (float)DIM)
               + 1.25f * accum[1] / ((float)N_TOK * (float)LAT)
               + 0.1f * L;
    out[0] = loss;
    out[1] = expf(L);
    out[2] = (float)U / (float)VCB;
  }
}

extern "C" void kernel_launch(void* const* d_in, const int* in_sizes, int n_in,
                              void* d_out, int out_size, void* d_ws, size_t ws_size,
                              hipStream_t stream) {
  const float* x        = (const float*)d_in[0];   // [8,1024,1024]
  const float* codebook = (const float*)d_in[1];   // [8192,1024]
  const float* W_in     = (const float*)d_in[2];   // [1024,512]
  const float* b_in     = (const float*)d_in[3];   // [512]
  const float* W_code   = (const float*)d_in[4];   // [1024,512]
  const float* b_code   = (const float*)d_in[5];   // [512]
  float* out = (float*)d_out;

  char* ws = (char*)d_ws;
  size_t off = 0;
  __bf16* WT_in   = (__bf16*)(ws + off); off += (size_t)LAT * DIM * 2;       // 1 MB
  __bf16* WT_code = (__bf16*)(ws + off); off += (size_t)LAT * DIM * 2;       // 1 MB
  float*  latx    = (float*)(ws + off);  off += (size_t)N_TOK * LAT * 4;     // 16 MB
  float*  latc    = (float*)(ws + off);  off += (size_t)VCB * LAT * 4;       // 16 MB
  __bf16* abf     = (__bf16*)(ws + off); off += (size_t)N_TOK * LAT * 2;     // 8 MB
  __bf16* cbf     = (__bf16*)(ws + off); off += (size_t)VCB * LAT * 2;       // 8 MB
  unsigned long long* packed = (unsigned long long*)(ws + off); off += (size_t)N_TOK * 8;
  int*    hist    = (int*)(ws + off);    off += (size_t)VCB * 4;
  float*  accum   = (float*)(ws + off);  off += 256;

  prep_w_kernel<<<(LAT * DIM) / 256, 256, 0, stream>>>(W_in, WT_in);
  prep_w_kernel<<<(LAT * DIM) / 256, 256, 0, stream>>>(W_code, WT_code);
  init_kernel<<<VCB / 256, 256, 0, stream>>>(packed, hist, accum);

  // latent projections (WMMA bf16)
  gemm_latent_kernel<<<128, 128, 0, stream>>>(x, WT_in, b_in, latx, nullptr, nullptr, 0);
  gemm_latent_kernel<<<128, 128, 0, stream>>>(codebook, WT_code, b_code, latc, nullptr, nullptr, 0);

  normalize_kernel<<<N_TOK / 8, 256, 0, stream>>>(latx, abf, N_TOK);
  normalize_kernel<<<VCB / 8, 256, 0, stream>>>(latc, cbf, VCB);

  // fused 8192x8192x512 similarity GEMM + argmax
  dist_argmax_kernel<<<256, 128, 0, stream>>>(abf, cbf, packed);

  // gather + straight-through output + sse1 + histogram
  gather_kernel<<<N_TOK, 256, 0, stream>>>(packed, codebook, x, out, hist, accum);

  // latent_quantized GEMM with fused sse2 reduction (A = quantized in d_out)
  gemm_latent_kernel<<<128, 128, 0, stream>>>(out, WT_code, b_code, nullptr, latx, accum + 1, 1);

  finalize_kernel<<<1, 256, 0, stream>>>(hist, accum, out + (size_t)N_TOK * DIM);
}
